// MoERouter_89524298318521
// MI455X (gfx1250) — matrix operand order repaired and checked by reading, not compiled, not run on hardware
//
#include <hip/hip_runtime.h>

typedef __attribute__((ext_vector_type(16))) __bf16          v16bf;
typedef __attribute__((ext_vector_type(16))) unsigned short  v16u;
typedef __attribute__((ext_vector_type(8)))  float           v8f;

#define BROWS   65536
#define DMODEL  2048
#define NEXP    64
#define TOPK    8
#define WG_ROWS 128
#define NWG     (BROWS / WG_ROWS)   /* 512 */
#define LSTRIDE 65                  /* padded LDS row stride (banks=64) */
#define AUXCOEF 0.01f

static __device__ __forceinline__ unsigned short f2bf(float f) {
    union { float f; unsigned u; } v; v.f = f;
    unsigned r = v.u + 0x7FFFu + ((v.u >> 16) & 1u);   // round-to-nearest-even
    return (unsigned short)(r >> 16);
}

// ---------------------------------------------------------------------------
// Kernel 0: repack W (DMODEL x NEXP, k-major f32) into bf16 WMMA B-fragments.
// Fragment f = kblock*4 + ntile covers K in [kblock*32, kblock*32+32),
// N in [ntile*16, ntile*16+16). Per-lane layout (contiguous k order used by this code, not the ISA table order):
// lanes 0-15: N = lane, K = kblock*32 + j (j=0..15)
// lanes 16-31: N = lane-16, K = kblock*32 + 16 + j
// Stored as 32 lanes x 16 ushort contiguous -> one 32B load per lane.
// ---------------------------------------------------------------------------
__global__ __launch_bounds__(256) void moe_pack_w(const float* __restrict__ W,
                                                  unsigned short* __restrict__ Wp) {
    int t = blockIdx.x * 256 + threadIdx.x;   // 8192 threads total
    int lane   = t & 31;
    int frag   = t >> 5;                      // 0..255
    int ntile  = frag & 3;
    int kblock = frag >> 2;
    int n      = ntile * 16 + (lane & 15);
    int kbase  = kblock * 32 + (lane >> 4) * 16;
    v16u out;
    #pragma unroll
    for (int j = 0; j < 16; ++j)
        out[j] = f2bf(W[(size_t)(kbase + j) * NEXP + n]);
    *(v16u*)(Wp + (size_t)frag * 512 + lane * 16) = out;
}

// ---------------------------------------------------------------------------
// Kernel 1: fused router. 8 waves/WG, 16 rows/wave => 128 rows/WG, 512 WGs.
// ---------------------------------------------------------------------------
__global__ __launch_bounds__(256) void moe_router_main(
        const float* __restrict__ x,
        const unsigned short* __restrict__ Wp,
        const float* __restrict__ ebias,
        float* __restrict__ gates,
        float* __restrict__ idx_out,
        float* __restrict__ Ppart,
        float* __restrict__ Fpart)
{
    __shared__ float s_log[WG_ROWS * LSTRIDE];
    __shared__ float s_rmax[WG_ROWS];
    __shared__ float s_rsum[WG_ROWS];
    __shared__ float s_bias[NEXP];
    __shared__ float s_f[NEXP];

    const int tid  = threadIdx.x;
    const int lane = tid & 31;
    const int wave = tid >> 5;

    if (tid < NEXP) { s_bias[tid] = ebias[tid]; s_f[tid] = 0.0f; }

    // ---- Phase 1: WMMA GEMM, each wave -> 16 rows x 64 experts ----
    // A fragment lane map (16-bit A 16x32): lanes 0-15 row=lane, K{0..15};
    // lanes 16-31 row=lane-16, K{16..31}.
    const int    r    = lane & 15;
    const int    kg   = lane >> 4;
    const size_t grow = (size_t)blockIdx.x * WG_ROWS + wave * 16 + r;
    const float* xrow = x + grow * DMODEL;

    v8f acc0 = {}, acc1 = {}, acc2 = {}, acc3 = {};

    for (int kb = 0; kb < DMODEL; kb += 32) {
        const float4* ap = (const float4*)(xrow + kb + kg * 16);
        float4 f0 = ap[0], f1 = ap[1];   // K = kb + kg*16 + {0..7}
        float4 f2 = ap[2], f3 = ap[3];   // K = kb + kg*16 + {8..15}
        v16u au;
        au[0]=f2bf(f0.x);  au[1]=f2bf(f0.y);  au[2]=f2bf(f0.z);  au[3]=f2bf(f0.w);
        au[4]=f2bf(f1.x);  au[5]=f2bf(f1.y);  au[6]=f2bf(f1.z);  au[7]=f2bf(f1.w);
        au[8]=f2bf(f2.x);  au[9]=f2bf(f2.y);  au[10]=f2bf(f2.z); au[11]=f2bf(f2.w);
        au[12]=f2bf(f3.x); au[13]=f2bf(f3.y); au[14]=f2bf(f3.z); au[15]=f2bf(f3.w);
        v16bf a = __builtin_bit_cast(v16bf, au);

        const unsigned short* bbase = Wp + ((size_t)(kb >> 5) * 4) * 512 + lane * 16;
        v16bf b0 = __builtin_bit_cast(v16bf, *(const v16u*)(bbase));
        v16bf b1 = __builtin_bit_cast(v16bf, *(const v16u*)(bbase + 512));
        v16bf b2 = __builtin_bit_cast(v16bf, *(const v16u*)(bbase + 1024));
        v16bf b3 = __builtin_bit_cast(v16bf, *(const v16u*)(bbase + 1536));

        acc0 = __builtin_amdgcn_wmma_f32_16x16x32_bf16(false, a, false, b0, (short)0, acc0, false, false);
        acc1 = __builtin_amdgcn_wmma_f32_16x16x32_bf16(false, a, false, b1, (short)0, acc1, false, false);
        acc2 = __builtin_amdgcn_wmma_f32_16x16x32_bf16(false, a, false, b2, (short)0, acc2, false, false);
        acc3 = __builtin_amdgcn_wmma_f32_16x16x32_bf16(false, a, false, b3, (short)0, acc3, false, false);
    }

    // C/D layout: lanes 0-15: M=vgpr, N=lane ; lanes 16-31: M=vgpr+8, N=lane-16
    const int m_off = (lane < 16) ? 0 : 8;
    const int col0  = lane & 15;
    const int rbase = wave * 16;
    #pragma unroll
    for (int rr = 0; rr < 8; ++rr) {
        float* p = &s_log[(rbase + rr + m_off) * LSTRIDE + col0];
        p[0]  = acc0[rr];
        p[16] = acc1[rr];
        p[32] = acc2[rr];
        p[48] = acc3[rr];
    }
    __syncthreads();

    // ---- Phase 2a: per-row top-8 (biased) + softmax over unbiased selected ----
    float gsel[TOPK];
    int   isel[TOPK];
    if (tid < WG_ROWS) {
        const float* rowp = &s_log[tid * LSTRIDE];
        unsigned long long taken = 0ull;
        float vals[TOPK];
        #pragma unroll
        for (int k = 0; k < TOPK; ++k) {
            float best = -__builtin_inff();
            int   bn   = 0;
            for (int n = 0; n < NEXP; ++n) {
                if ((taken >> n) & 1ull) continue;
                float v = rowp[n] + s_bias[n];
                if (v > best) { best = v; bn = n; }
            }
            taken |= (1ull << bn);
            isel[k] = bn;
            vals[k] = rowp[bn];          // UNbiased logit
        }
        float m = vals[0];
        #pragma unroll
        for (int k = 1; k < TOPK; ++k) m = fmaxf(m, vals[k]);
        float s = 0.0f;
        #pragma unroll
        for (int k = 0; k < TOPK; ++k) { gsel[k] = __expf(vals[k] - m); s += gsel[k]; }
        float inv = 1.0f / s;
        #pragma unroll
        for (int k = 0; k < TOPK; ++k) gsel[k] *= inv;

        // full-row softmax stats for P_i
        float rm = -__builtin_inff();
        for (int n = 0; n < NEXP; ++n) rm = fmaxf(rm, rowp[n]);
        float rs = 0.0f;
        for (int n = 0; n < NEXP; ++n) rs += __expf(rowp[n] - rm);
        s_rmax[tid] = rm;
        s_rsum[tid] = 1.0f / rs;

        #pragma unroll
        for (int k = 0; k < TOPK; ++k) atomicAdd(&s_f[isel[k]], 1.0f);

        size_t rg = (size_t)blockIdx.x * WG_ROWS + tid;
        #pragma unroll
        for (int k = 0; k < TOPK; ++k) idx_out[rg * TOPK + k] = (float)isel[k];
    }
    __syncthreads();

    // ---- Phase 2b: per-WG deterministic partials for P_i / f_i ----
    if (tid < NEXP) {
        float accp = 0.0f;
        for (int rr = 0; rr < WG_ROWS; ++rr)
            accp += __expf(s_log[rr * LSTRIDE + tid] - s_rmax[rr]) * s_rsum[rr];
        Ppart[(size_t)blockIdx.x * NEXP + tid] = accp;
        Fpart[(size_t)blockIdx.x * NEXP + tid] = s_f[tid];
    }
    __syncthreads();

    // ---- Phase 2c: rebuild sparse gate rows in LDS ----
    if (tid < WG_ROWS) {
        float* rowp = &s_log[tid * LSTRIDE];
        for (int n = 0; n < NEXP; ++n) rowp[n] = 0.0f;
        #pragma unroll
        for (int k = 0; k < TOPK; ++k) rowp[isel[k]] = gsel[k];
    }
    __syncthreads();

    // ---- Phase 2d: coalesced gate store ----
    for (int i = tid; i < WG_ROWS * NEXP; i += 256) {
        int rr = i >> 6, cc = i & 63;
        gates[((size_t)blockIdx.x * WG_ROWS + rr) * NEXP + cc] = s_log[rr * LSTRIDE + cc];
    }
}

// ---------------------------------------------------------------------------
// Kernel 2: deterministic reduction of per-WG partials -> aux loss scalar
// ---------------------------------------------------------------------------
__global__ __launch_bounds__(64) void moe_finalize(const float* __restrict__ Ppart,
                                                   const float* __restrict__ Fpart,
                                                   float* __restrict__ aux) {
    __shared__ float s[NEXP];
    int n = threadIdx.x;   // 64 threads
    float ps = 0.0f, fs = 0.0f;
    for (int w = 0; w < NWG; ++w) {
        ps += Ppart[(size_t)w * NEXP + n];
        fs += Fpart[(size_t)w * NEXP + n];
    }
    s[n] = (fs * (1.0f / (float)BROWS)) * (ps * (1.0f / (float)BROWS));
    __syncthreads();
    if (n == 0) {
        float a = 0.0f;
        for (int i = 0; i < NEXP; ++i) a += s[i];
        *aux = AUXCOEF * (float)NEXP * a;
    }
}

extern "C" void kernel_launch(void* const* d_in, const int* in_sizes, int n_in,
                              void* d_out, int out_size, void* d_ws, size_t ws_size,
                              hipStream_t stream) {
    const float* x  = (const float*)d_in[0];            // (65536, 2048) f32
    const float* W  = (const float*)d_in[1];            // (2048, 64) f32
    const float* eb = (const float*)d_in[2];            // (64,) f32

    float* gates  = (float*)d_out;                       // 65536*64
    float* idx_o  = gates + (size_t)BROWS * NEXP;        // 65536*8 (as float)
    float* aux_o  = idx_o + (size_t)BROWS * TOPK;        // 1

    // workspace: Wp (256 KB bf16 fragments) | Ppart (512*64 f32) | Fpart (512*64 f32)
    unsigned short* Wp = (unsigned short*)d_ws;
    float* Ppart = (float*)((char*)d_ws + 256 * 1024);
    float* Fpart = Ppart + (size_t)NWG * NEXP;

    moe_pack_w<<<32, 256, 0, stream>>>(W, Wp);
    moe_router_main<<<NWG, 256, 0, stream>>>(x, Wp, eb, gates, idx_o, Ppart, Fpart);
    moe_finalize<<<1, 64, 0, stream>>>(Ppart, Fpart, aux_o);
}
